// GCNLSTM_36679020708494
// MI455X (gfx1250) — compile-verified, matrix operand order checked by reading, not run on hardware
//
#include <hip/hip_runtime.h>

// ---------------------------------------------------------------------------
// GCN(2 layers) + 2-layer LSTM + FC for MI455X (gfx1250), all GEMMs on
// v_wmma_f32_16x16x32_bf16 with f32 accumulation.
//   B=16 T=64 N=512 GH=H=64 BT=1024 SEQ=B*N=8192
// ---------------------------------------------------------------------------

typedef __attribute__((ext_vector_type(16))) __bf16        v16bf;
typedef __attribute__((ext_vector_type(8)))  float         v8f;
typedef __attribute__((ext_vector_type(4)))  unsigned int  v4u;

union Frag16 {
  v16bf v;
  unsigned int u[8];
  v4u q[2];
};

__device__ __forceinline__ v8f wmma_bf16(v16bf a, v16bf b, v8f c) {
  // (neg_a, A, neg_b, B, c_mod, C, reuse_a, reuse_b)
  return __builtin_amdgcn_wmma_f32_16x16x32_bf16(false, a, false, b, (short)0, c,
                                                 false, false);
}

// Gather a 16x32 bf16 A-fragment (or matching B-fragment from an N-major
// [col][k] tile) out of LDS. Tile stored row-major, k contiguous,
// `strideDw` dwords per row, `baseDw` selects the k-chunk.
// Per ISA layout: lane L: row = L&15, kh = L>>4; vector pair p holds
// k = (p>=4?16:0) + kh*8 + (p&3)*2 and k+1 (adjacent in memory -> b32 load).
__device__ __forceinline__ v16bf frag_gather(const unsigned int* lds, int row,
                                             int kh, int strideDw, int baseDw) {
  Frag16 f;
#pragma unroll
  for (int p = 0; p < 8; ++p) {
    const int k = ((p >> 2) << 4) + kh * 8 + ((p & 3) << 1);
    f.u[p] = lds[row * strideDw + baseDw + (k >> 1)];
  }
  return f.v;
}

__device__ __forceinline__ unsigned int pack_bf2(float a, float b) {
  union { __bf16 h[2]; unsigned int u; } p;
  p.h[0] = (__bf16)a;
  p.h[1] = (__bf16)b;
  return p.u;
}

__device__ __forceinline__ float sigm(float x) {
  return 1.0f / (1.0f + __expf(-x));
}
__device__ __forceinline__ float tanh_(float x) {
  x = fminf(fmaxf(x, -15.f), 15.f);
  const float e = __expf(2.0f * x);
  return (e - 1.0f) / (e + 1.0f);
}

// ---------------------------------------------------------------------------
// f32 -> bf16 copy
// ---------------------------------------------------------------------------
__global__ void cvt_bf16_k(const float* __restrict__ src, __bf16* __restrict__ dst,
                           int n) {
  const int i = blockIdx.x * blockDim.x + threadIdx.x;
  if (i < n) dst[i] = (__bf16)src[i];
}

// ---------------------------------------------------------------------------
// GCN layer 1: S[n,bt] = adj[n,:] . X[bt,:]  (512x512x1024 GEMM)
// h1[bt,n,g] = relu(W1[g]*S + b1[g]); one wave per 16x16 output tile.
// ---------------------------------------------------------------------------
__global__ __launch_bounds__(32) void gcn1_k(const __bf16* __restrict__ adjbf,
                                             const __bf16* __restrict__ xbf,
                                             const float* __restrict__ W1,
                                             const float* __restrict__ b1,
                                             __bf16* __restrict__ h1) {
  __shared__ unsigned int ldsA[256];   // adj tile 16x32 bf16
  __shared__ unsigned int ldsB[256];   // X tile  [bt 16][k 32] bf16 (== B N-major)
  __shared__ float w1s[64];
  __shared__ float b1s[64];

  const int lane = threadIdx.x;
  const int btt = blockIdx.x;   // 0..63
  const int nt  = blockIdx.y;   // 0..31
  const int n0 = nt * 16, bt0 = btt * 16;

  w1s[lane]      = W1[lane];
  w1s[lane + 32] = W1[lane + 32];
  b1s[lane]      = b1[lane];
  b1s[lane + 32] = b1[lane + 32];

  const int row = lane & 15, kh = lane >> 4;
  const unsigned int* gA = (const unsigned int*)adjbf;  // 256 dw per row
  const unsigned int* gB = (const unsigned int*)xbf;    // 256 dw per row

  v8f acc = {};
  for (int k0 = 0; k0 < 512; k0 += 32) {
    __syncthreads();
#pragma unroll
    for (int i = 0; i < 8; ++i) {
      const int f = i * 32 + lane;
      const int r = f >> 4, dw = f & 15;
      ldsA[f] = gA[(n0 + r) * 256 + (k0 >> 1) + dw];
      ldsB[f] = gB[(bt0 + r) * 256 + (k0 >> 1) + dw];
    }
    __syncthreads();
    const v16bf a = frag_gather(ldsA, row, kh, 16, 0);
    const v16bf b = frag_gather(ldsB, row, kh, 16, 0);
    acc = wmma_bf16(a, b, acc);
  }

  // C layout: element (M = r + 8*(lane>>4), N = lane&15); M=n row, N=bt col.
  const int col = lane & 15, hi = lane >> 4;
  const int bt = bt0 + col;
#pragma unroll
  for (int r = 0; r < 8; ++r) {
    const int n = n0 + r + 8 * hi;
    const float s = acc[r];
    v4u* dst = (v4u*)(h1 + ((size_t)bt * 512 + n) * 64);
#pragma unroll
    for (int gq = 0; gq < 8; ++gq) {
      v4u pk;
#pragma unroll
      for (int j = 0; j < 4; ++j) {
        const int g = gq * 8 + j * 2;
        const float v0 = fmaxf(fmaf(w1s[g],     s, b1s[g]),     0.f);
        const float v1 = fmaxf(fmaf(w1s[g + 1], s, b1s[g + 1]), 0.f);
        pk[j] = pack_bf2(v0, v1);
      }
      dst[gq] = pk;
    }
  }
}

// ---------------------------------------------------------------------------
// GCN layer 2, fused: A2 = adj(512x512) @ h1[bt](512x64);
// h2 = relu(A2 @ W2^T + b2), written directly in LSTM sequence layout
// seq[s=b*512+n][t][h]. One wave per (bt, 16-row tile).
// ---------------------------------------------------------------------------
__global__ __launch_bounds__(32) void gcn2_k(const __bf16* __restrict__ adjbf,
                                             const __bf16* __restrict__ h1,
                                             const __bf16* __restrict__ w2bf,
                                             const float* __restrict__ b2,
                                             __bf16* __restrict__ seq) {
  __shared__ unsigned int ldsA[256];    // adj tile 16x32
  __shared__ unsigned int ldsBt[1024];  // h1 tile transposed: [g 64][m 32] bf16
  __shared__ unsigned int ldsW2[2048];  // W2 [h 64][g 64] bf16 (N-major B layout)
  __shared__ unsigned int ldsA2[512];   // A2 staged as [m 16][g 64] bf16

  const int lane  = threadIdx.x;
  const int ntile = blockIdx.x;  // 0..31
  const int bt    = blockIdx.y;  // 0..1023
  const int n0 = ntile * 16;
  const int row = lane & 15, kh = lane >> 4;

  for (int i = lane; i < 2048; i += 32)
    ldsW2[i] = ((const unsigned int*)w2bf)[i];

  const unsigned int* gA = (const unsigned int*)adjbf;
  v8f acc[4] = {};

  for (int m0 = 0; m0 < 512; m0 += 32) {
    __syncthreads();
#pragma unroll
    for (int i = 0; i < 8; ++i) {
      const int f = i * 32 + lane;
      const int r = f >> 4, dw = f & 15;
      ldsA[f] = gA[(n0 + r) * 256 + (m0 >> 1) + dw];
    }
    {
      const __bf16* src = h1 + ((size_t)bt * 512 + m0) * 64;
      __bf16* btl = (__bf16*)ldsBt;
      for (int i = lane; i < 2048; i += 32) {
        const int mm = i >> 6, g = i & 63;      // coalesced reads over g
        btl[g * 32 + mm] = src[mm * 64 + g];    // transpose -> k contiguous
      }
    }
    if (m0 + 32 < 512)   // pull next adj tile toward the WGP (global_prefetch_b8)
      __builtin_prefetch(adjbf + (size_t)(n0 + row) * 512 + m0 + 32, 0, 1);
    __syncthreads();

    const v16bf a = frag_gather(ldsA, row, kh, 16, 0);
#pragma unroll
    for (int gt = 0; gt < 4; ++gt) {
      const v16bf b = frag_gather(ldsBt, gt * 16 + row, kh, 16, 0);
      acc[gt] = wmma_bf16(a, b, acc[gt]);
    }
  }

  // Relayout A2 accs into an A-fragment via LDS (C layout -> [m][g] bf16).
  const int col = lane & 15, hi = lane >> 4;
  __syncthreads();
  {
    __bf16* a2 = (__bf16*)ldsA2;
#pragma unroll
    for (int gt = 0; gt < 4; ++gt)
#pragma unroll
      for (int r = 0; r < 8; ++r)
        a2[(r + 8 * hi) * 64 + gt * 16 + col] = (__bf16)acc[gt][r];
  }
  __syncthreads();

  v8f d[4] = {};
#pragma unroll
  for (int kc = 0; kc < 2; ++kc) {
    const v16bf a = frag_gather(ldsA2, row, kh, 32, kc * 16);
#pragma unroll
    for (int ht = 0; ht < 4; ++ht) {
      const v16bf b = frag_gather(ldsW2, ht * 16 + row, kh, 32, kc * 16);
      d[ht] = wmma_bf16(a, b, d[ht]);
    }
  }

  const int bb = bt >> 6, tt = bt & 63;  // bt = b*T + t
#pragma unroll
  for (int ht = 0; ht < 4; ++ht)
#pragma unroll
    for (int r = 0; r < 8; ++r) {
      const int m = r + 8 * hi;
      const int h = ht * 16 + col;
      const float v = fmaxf(d[ht][r] + b2[h], 0.f);
      const size_t s = (size_t)bb * 512 + (n0 + m);
      seq[(s * 64 + tt) * 64 + h] = (__bf16)v;
    }
}

// ---------------------------------------------------------------------------
// One LSTM layer. Block = 4 waves; each wave owns 16 independent sequences.
// Weights (Wih, Whh as K x 4H B-matrices) are pre-gathered once into LDS in
// fragment-linear order -> hot loop is 64 wmma + b128 LDS reads per timestep.
// Gate trick: i/f/g/o for a given (seq,h) live in the same lane & acc slot of
// tiles q, q+4, q+8, q+12, so cell state c stays in 32 VGPRs all 64 steps.
// ---------------------------------------------------------------------------
__global__ __launch_bounds__(128) void lstm_k(const __bf16* __restrict__ inseq,
                                              const __bf16* __restrict__ wih,
                                              const __bf16* __restrict__ whh,
                                              const float* __restrict__ bih,
                                              const float* __restrict__ bhh,
                                              __bf16* __restrict__ outseq,
                                              float* __restrict__ outlast) {
  extern __shared__ unsigned int lds[];
  unsigned int* wfrag = lds;                  // 64 frags * 256 dw = 16384 dw (64KB)
  unsigned int* hbuf  = lds + 16384;          // 4 waves * 512 dw
  unsigned int* xbuf  = lds + 16384 + 2048;   // 4 waves * 512 dw

  const int tid  = threadIdx.x;
  const int lane = tid & 31;
  const int wave = tid >> 5;
  const int row = lane & 15, kh = lane >> 4;
  const int col = row, hi = kh;
  const int seq0 = blockIdx.x * 64 + wave * 16;

  // Pre-gather weight B-fragments (frag f: w=f>>5, nt=(f>>1)&15, kc=f&1).
#pragma unroll
  for (int fi = 0; fi < 16; ++fi) {
    const int f = wave * 16 + fi;
    const int w = f >> 5;
    const int ntt = (f >> 1) & 15;
    const int kc = f & 1;
    const unsigned int* Wsrc = (const unsigned int*)(w ? whh : wih); // [256][32 dw]
    const int n = ntt * 16 + row;
#pragma unroll
    for (int p = 0; p < 8; ++p) {
      const int k = kc * 32 + ((p >> 2) << 4) + kh * 8 + ((p & 3) << 1);
      wfrag[f * 256 + lane * 8 + p] = Wsrc[n * 32 + (k >> 1)];
    }
  }

  float bsum[16];
#pragma unroll
  for (int ntt = 0; ntt < 16; ++ntt)
    bsum[ntt] = bih[ntt * 16 + col] + bhh[ntt * 16 + col];

  unsigned int* myh = hbuf + wave * 512;
  unsigned int* myx = xbuf + wave * 512;
  for (int i = lane; i < 512; i += 32) myh[i] = 0u;

  float c[4][8];
#pragma unroll
  for (int q = 0; q < 4; ++q)
#pragma unroll
    for (int r = 0; r < 8; ++r) c[q][r] = 0.f;

  __syncthreads();  // weights + h0 visible

  const unsigned int* gx = (const unsigned int*)inseq;
  for (int t = 0; t < 64; ++t) {
    // Stage x_t for our 16 sequences: [j 16][h 64] bf16 = 512 dw.
#pragma unroll
    for (int i = 0; i < 16; ++i) {
      const int f = i * 32 + lane;
      const int j = f >> 5, dw = f & 31;
      myx[j * 32 + dw] = gx[((size_t)(seq0 + j) * 64 + t) * 32 + dw];
    }
    const v16bf ax0 = frag_gather(myx, row, kh, 32, 0);
    const v16bf ax1 = frag_gather(myx, row, kh, 32, 16);
    const v16bf ah0 = frag_gather(myh, row, kh, 32, 0);
    const v16bf ah1 = frag_gather(myh, row, kh, 32, 16);

    v8f acc[16];
#pragma unroll
    for (int ntt = 0; ntt < 16; ++ntt)
#pragma unroll
      for (int r = 0; r < 8; ++r) acc[ntt][r] = bsum[ntt];

#pragma unroll
    for (int ntt = 0; ntt < 16; ++ntt) {
#pragma unroll
      for (int kc = 0; kc < 2; ++kc) {
        Frag16 bw;
        const v4u* q1 = (const v4u*)(wfrag + (ntt * 2 + kc) * 256 + lane * 8);
        bw.q[0] = q1[0]; bw.q[1] = q1[1];
        acc[ntt] = wmma_bf16(kc ? ax1 : ax0, bw.v, acc[ntt]);
        const v4u* q2 = (const v4u*)(wfrag + (32 + ntt * 2 + kc) * 256 + lane * 8);
        bw.q[0] = q2[0]; bw.q[1] = q2[1];
        acc[ntt] = wmma_bf16(kc ? ah1 : ah0, bw.v, acc[ntt]);
      }
    }

    __bf16* hb = (__bf16*)myh;
#pragma unroll
    for (int q = 0; q < 4; ++q) {
#pragma unroll
      for (int r = 0; r < 8; ++r) {
        const float iv = acc[q][r];
        const float fv = acc[q + 4][r];
        const float gv = acc[q + 8][r];
        const float ov = acc[q + 12][r];
        const float cc = sigm(fv) * c[q][r] + sigm(iv) * tanh_(gv);
        c[q][r] = cc;
        const float hh = sigm(ov) * tanh_(cc);
        const int m = r + 8 * hi;
        const int h = q * 16 + col;
        const __bf16 hv = (__bf16)hh;
        hb[m * 64 + h] = hv;  // per-wave LDS, in-order with next gather
        outseq[((size_t)(seq0 + m) * 64 + t) * 64 + h] = hv;
        if (t == 63) outlast[(size_t)(seq0 + m) * 64 + h] = hh;
      }
    }
  }
}

// ---------------------------------------------------------------------------
// Final FC: out[s] = fc_b + last[s,:] . fc_W[0,:]
// ---------------------------------------------------------------------------
__global__ void fc_k(const float* __restrict__ last, const float* __restrict__ fcW,
                     const float* __restrict__ fcb, float* __restrict__ out) {
  const int s = blockIdx.x * blockDim.x + threadIdx.x;
  if (s >= 8192) return;
  float a = fcb[0];
#pragma unroll
  for (int h = 0; h < 64; ++h) a = fmaf(last[(size_t)s * 64 + h], fcW[h], a);
  out[s] = a;
}

// ---------------------------------------------------------------------------
extern "C" void kernel_launch(void* const* d_in, const int* in_sizes, int n_in,
                              void* d_out, int out_size, void* d_ws, size_t ws_size,
                              hipStream_t stream) {
  (void)in_sizes; (void)n_in; (void)out_size; (void)ws_size;
  const float* X    = (const float*)d_in[0];
  const float* adj  = (const float*)d_in[1];
  const float* g1W  = (const float*)d_in[2];
  const float* g1b  = (const float*)d_in[3];
  const float* g2W  = (const float*)d_in[4];
  const float* g2b  = (const float*)d_in[5];
  const float* Wih0 = (const float*)d_in[6];
  const float* Whh0 = (const float*)d_in[7];
  const float* bih0 = (const float*)d_in[8];
  const float* bhh0 = (const float*)d_in[9];
  const float* Wih1 = (const float*)d_in[10];
  const float* Whh1 = (const float*)d_in[11];
  const float* bih1 = (const float*)d_in[12];
  const float* bhh1 = (const float*)d_in[13];
  const float* fcW  = (const float*)d_in[14];
  const float* fcb  = (const float*)d_in[15];
  float* out = (float*)d_out;

  // Workspace layout (needs ~138 MB):
  char* ws = (char*)d_ws;
  __bf16* adjbf = (__bf16*)(ws + 0);                      // 512*512*2   = 512 KB
  __bf16* xbf   = (__bf16*)(ws + 524288);                 // 1024*512*2  = 1 MB
  __bf16* w2bf  = (__bf16*)(ws + 1572864);                // 64*64*2
  __bf16* wih0b = (__bf16*)(ws + 1581056);                // 256*64*2
  __bf16* whh0b = (__bf16*)(ws + 1613824);
  __bf16* wih1b = (__bf16*)(ws + 1646592);
  __bf16* whh1b = (__bf16*)(ws + 1679360);
  __bf16* bufA  = (__bf16*)(ws + 1712128);                // 64 MB ping
  __bf16* bufB  = (__bf16*)(ws + 1712128 + 67108864);     // 64 MB pong
  float*  lastb = (float*)(ws + 1712128 + 2 * 67108864);  // 8192*64*4 = 2 MB

  auto cvt = [&](const float* s, __bf16* d, int n) {
    cvt_bf16_k<<<(n + 255) / 256, 256, 0, stream>>>(s, d, n);
  };
  cvt(adj, adjbf, 512 * 512);
  cvt(X, xbf, 1024 * 512);           // X is [bt][m] contiguous already
  cvt(g2W, w2bf, 64 * 64);
  cvt(Wih0, wih0b, 256 * 64);
  cvt(Whh0, whh0b, 256 * 64);
  cvt(Wih1, wih1b, 256 * 64);
  cvt(Whh1, whh1b, 256 * 64);

  gcn1_k<<<dim3(64, 32), 32, 0, stream>>>(adjbf, xbf, g1W, g1b, bufA);
  gcn2_k<<<dim3(32, 1024), 32, 0, stream>>>(adjbf, bufA, w2bf, g2b, bufB);

  const size_t lstm_lds = (16384 + 2048 + 2048) * sizeof(unsigned int); // 80 KB
  lstm_k<<<128, 128, lstm_lds, stream>>>(bufB, wih0b, whh0b, bih0, bhh0, bufA, lastb);
  lstm_k<<<128, 128, lstm_lds, stream>>>(bufA, wih1b, whh1b, bih1, bhh1, bufB, lastb);

  fc_k<<<32, 256, 0, stream>>>(lastb, fcW, fcb, out);
}
